// MambaBlock_17102559773097
// MI455X (gfx1250) — matrix operand order repaired and checked by reading, not compiled, not run on hardware
//
#include <hip/hip_runtime.h>
#include <hip/hip_bf16.h>

// ---------------------------------------------------------------------------
// Mamba2 block for MI455X (gfx1250, wave32, WMMA + async-to-LDS staging).
// Pipeline: LN -> bf16 WMMA GEMM (in_proj) -> dt/dA -> conv4+SiLU ->
//           register-resident selective scan -> gate+RMS -> bf16 WMMA GEMM
//           (out_proj, fused +x residual).
// GEMM: wave tile 64x32 (8 v_wmma / K-step). B panel (shared by all 8 waves)
// is double-buffered in LDS via global_load_async_to_lds_b128 + s_wait_asynccnt;
// A is streamed from global with global_prefetch.
// ---------------------------------------------------------------------------

typedef __attribute__((ext_vector_type(16))) __bf16 v16bf;
typedef __attribute__((ext_vector_type(8)))  __bf16 v8bf;
typedef __attribute__((ext_vector_type(8)))  float  v8f;

#define B_    4
#define L_    2048
#define DM    1024
#define DST   128
#define DCONV 4
#define DIN   2048
#define NH    32
#define HD    64
#define CONVD (DIN + 2*DST)          // 2304
#define DPROJ (2*DIN + 2*DST + NH)   // 4384
#define ROWS  (B_*L_)                // 8192
#define EPS_  1e-5f

// ------------------------------ utility ------------------------------------

__global__ void f32_to_bf16_kernel(const float* __restrict__ in,
                                   __bf16* __restrict__ out, int n) {
    int i = blockIdx.x * blockDim.x + threadIdx.x;
    if (i < n) out[i] = (__bf16)in[i];
}

// ------------------------------ LayerNorm ----------------------------------
__global__ __launch_bounds__(256)
void ln_kernel(const float* __restrict__ x, const float* __restrict__ w,
               const float* __restrict__ b, __bf16* __restrict__ u) {
    int row = blockIdx.x;
    const float* xr = x + (size_t)row * DM;
    float s = 0.f, s2 = 0.f;
    for (int i = threadIdx.x; i < DM; i += 256) {
        float v = xr[i]; s += v; s2 += v * v;
    }
    #pragma unroll
    for (int m = 16; m; m >>= 1) {
        s  += __shfl_xor(s,  m, 32);
        s2 += __shfl_xor(s2, m, 32);
    }
    __shared__ float ss[8], ss2[8];
    if ((threadIdx.x & 31) == 0) { ss[threadIdx.x >> 5] = s; ss2[threadIdx.x >> 5] = s2; }
    __syncthreads();
    s = 0.f; s2 = 0.f;
    #pragma unroll
    for (int i = 0; i < 8; i++) { s += ss[i]; s2 += ss2[i]; }
    float mu  = s * (1.f / DM);
    float var = s2 * (1.f / DM) - mu * mu;
    float r   = rsqrtf(var + EPS_);
    __bf16* ur = u + (size_t)row * DM;
    for (int i = threadIdx.x; i < DM; i += 256)
        ur[i] = (__bf16)(((xr[i] - mu) * r) * w[i] + b[i]);
}

// ------------------------------ WMMA GEMM ----------------------------------
// C[M,N] = A[M,K] * W[N,K]^T (+ resid), A/W bf16 row-major, C f32.
// Block: 256 thr = 8 waves stacked in M; block tile 512(M) x 32(N);
// wave tile 64 x 32 = 4 A-frags x 2 B-frags -> 8 accumulators.
// B panel staged to LDS asynchronously (double-buffered K chunks of 256).
#define KCHUNK 256
#define BPITCH 264   // 256 + 8 bf16 pad -> row stride 528B (bank stride 4)

template<int KDIM>
__global__ __launch_bounds__(256)
void gemm_bf16_kernel(const __bf16* __restrict__ A,
                      const __bf16* __restrict__ W,
                      float* __restrict__ C,
                      const float* __restrict__ resid,
                      int N) {
    const int lane = threadIdx.x & 31;
    const int wid  = threadIdx.x >> 5;
    const int half = lane >> 4;
    const int lr   = lane & 15;
    const int m0   = blockIdx.x * 512 + wid * 64;
    const int n0   = blockIdx.y * 32;

    __shared__ __align__(16) __bf16 bsh[2][32 * BPITCH];

    const __bf16* Arf[4];
    #pragma unroll
    for (int f = 0; f < 4; f++)
        Arf[f] = A + (size_t)(m0 + 16 * f + lr) * KDIM;

    // stage one K-chunk of the 32-row B panel into LDS (async, no VGPR data)
    auto stage = [&](int chunk, int buf) {
        const __bf16* Wbase = W + (size_t)n0 * KDIM + (size_t)chunk * KCHUNK;
        #pragma unroll
        for (int i = 0; i < 4; i++) {
            int j  = threadIdx.x + i * 256;   // 0..1023 transfers of 16B
            int r  = j >> 5;                  // B row 0..31
            int kk = (j & 31) << 3;           // K element 0..248 step 8
            const __bf16* src = Wbase + (size_t)r * KDIM + kk;
            unsigned dst = (unsigned)(size_t)(void*)&bsh[buf][r * BPITCH + kk];
            asm volatile("global_load_async_to_lds_b128 %0, %1, off"
                         :: "v"(dst), "v"(src) : "memory");
        }
    };

    v8f acc[4][2] = {};
    union bfv { v16bf v; v8bf h[2]; };

    constexpr int NC = KDIM / KCHUNK;
    stage(0, 0);
    for (int c = 0; c < NC; c++) {
        __syncthreads();                       // prev chunk's readers done
        if (c + 1 < NC) {
            stage(c + 1, (c + 1) & 1);
            asm volatile("s_wait_asynccnt 0x4" ::: "memory");  // chunk c done
        } else {
            asm volatile("s_wait_asynccnt 0x0" ::: "memory");
        }
        __syncthreads();                       // chunk c visible to all waves
        const __bf16* bb = &bsh[c & 1][0];

        for (int kl = 0; kl < KCHUNK; kl += 32) {
            const int k0 = c * KCHUNK + kl;
            bfv b0, b1;
            b0.h[0] = *(const v8bf*)(bb + lr * BPITCH + kl + half * 16);
            b0.h[1] = *(const v8bf*)(bb + lr * BPITCH + kl + half * 16 + 8);
            b1.h[0] = *(const v8bf*)(bb + (lr + 16) * BPITCH + kl + half * 16);
            b1.h[1] = *(const v8bf*)(bb + (lr + 16) * BPITCH + kl + half * 16 + 8);
            #pragma unroll
            for (int f = 0; f < 4; f++) {
                bfv a;
                a.h[0] = *(const v8bf*)(Arf[f] + k0 + half * 16);
                a.h[1] = *(const v8bf*)(Arf[f] + k0 + half * 16 + 8);
                if (k0 + 128 < KDIM)
                    __builtin_prefetch(Arf[f] + k0 + 128 + half * 8, 0, 1);
                acc[f][0] = __builtin_amdgcn_wmma_f32_16x16x32_bf16(
                    false, a.v, false, b0.v, (short)0, acc[f][0], false, false);
                acc[f][1] = __builtin_amdgcn_wmma_f32_16x16x32_bf16(
                    false, a.v, false, b1.v, (short)0, acc[f][1], false, false);
            }
        }
    }

    #pragma unroll
    for (int f = 0; f < 4; f++) {
        #pragma unroll
        for (int i = 0; i < 8; i++) {
            int m = m0 + 16 * f + half * 8 + i;
            size_t o0 = (size_t)m * N + n0 + lr;
            float r0 = acc[f][0][i], r1 = acc[f][1][i];
            if (resid) { r0 += resid[o0]; r1 += resid[o0 + 16]; }
            C[o0] = r0; C[o0 + 16] = r1;
        }
    }
}

// ------------------------------ dt / dA ------------------------------------
__global__ void dt_kernel(const float* __restrict__ zx,
                          const float* __restrict__ dt_bias,
                          const float* __restrict__ A_log,
                          float* __restrict__ dt, float* __restrict__ dA) {
    int i = blockIdx.x * blockDim.x + threadIdx.x;   // over ROWS*NH
    if (i >= ROWS * NH) return;
    int h   = i & (NH - 1);
    int row = i >> 5;                                // NH == 32
    float v  = zx[(size_t)row * DPROJ + (DPROJ - NH) + h] + dt_bias[h];
    float sp = (v > 20.f) ? v : log1pf(expf(v));     // softplus
    float a  = -expf(A_log[h]);
    dt[i] = sp;
    dA[i] = expf(sp * a);
}

// ------------------------------ conv4 + SiLU -------------------------------
__global__ void conv_kernel(const float* __restrict__ zx,
                            const float* __restrict__ cw,
                            const float* __restrict__ cb,
                            float* __restrict__ out) {
    int i = blockIdx.x * blockDim.x + threadIdx.x;   // over ROWS*CONVD
    if (i >= ROWS * CONVD) return;
    int c   = i % CONVD;
    int row = i / CONVD;                             // b*L + l
    int l   = row % L_;
    float acc = cb[c];
    #pragma unroll
    for (int k = 0; k < DCONV; k++) {
        int t = l + k - (DCONV - 1);
        if (t >= 0)
            acc += cw[c * DCONV + k] * zx[(size_t)(row + k - (DCONV - 1)) * DPROJ + DIN + c];
    }
    out[i] = acc / (1.f + expf(-acc));               // SiLU
}

// ------------------------------ selective scan -----------------------------
__global__ __launch_bounds__(256)
void scan_kernel(const float* __restrict__ xbc,   // [ROWS, CONVD] post-conv
                 const float* __restrict__ dt,    // [ROWS, NH]
                 const float* __restrict__ dA,    // [ROWS, NH]
                 const float* __restrict__ Dp,    // [NH]
                 float* __restrict__ y) {         // [ROWS, DIN]
    int blk  = blockIdx.x;
    int b    = blk / NH, hh = blk % NH;
    int lane = threadIdx.x & 31, wid = threadIdx.x >> 5;
    int pbase = wid * 8;
    int nbase = lane * 4;

    float h[8][4];
    #pragma unroll
    for (int p = 0; p < 8; p++)
        #pragma unroll
        for (int j = 0; j < 4; j++) h[p][j] = 0.f;

    float Dh = Dp[hh];
    for (int t = 0; t < L_; t++) {
        size_t row = (size_t)b * L_ + t;
        const float* xr = xbc + row * CONVD;
        float4 Bv = *(const float4*)(xr + DIN + nbase);
        float4 Cv = *(const float4*)(xr + DIN + DST + nbase);
        float da  = dA[row * NH + hh];
        float dtv = dt[row * NH + hh];
        float xp[8];
        #pragma unroll
        for (int p = 0; p < 8; p++) xp[p] = xr[hh * HD + pbase + p];
        float* yout = y + row * DIN + hh * HD + pbase;
        #pragma unroll
        for (int p = 0; p < 8; p++) {
            float dx = dtv * xp[p];
            h[p][0] = da * h[p][0] + dx * Bv.x;
            h[p][1] = da * h[p][1] + dx * Bv.y;
            h[p][2] = da * h[p][2] + dx * Bv.z;
            h[p][3] = da * h[p][3] + dx * Bv.w;
            float part = h[p][0] * Cv.x + h[p][1] * Cv.y +
                         h[p][2] * Cv.z + h[p][3] * Cv.w;
            #pragma unroll
            for (int m = 16; m; m >>= 1) part += __shfl_xor(part, m, 32);
            if (lane == 0) yout[p] = part + Dh * xp[p];
        }
    }
}

// ------------------------------ gate + RMSNorm -----------------------------
__global__ __launch_bounds__(256)
void gate_kernel(const float* __restrict__ y, const float* __restrict__ zx,
                 const float* __restrict__ rms_w, __bf16* __restrict__ g) {
    int row = blockIdx.x;
    const float* yr = y  + (size_t)row * DIN;
    const float* zr = zx + (size_t)row * DPROJ;   // z = first DIN cols
    float vals[8];
    float s2 = 0.f;
    #pragma unroll
    for (int j = 0; j < 8; j++) {
        int i = threadIdx.x + j * 256;
        float z  = zr[i];
        float gv = yr[i] * (z / (1.f + expf(-z)));
        vals[j] = gv;
        s2 += gv * gv;
    }
    #pragma unroll
    for (int m = 16; m; m >>= 1) s2 += __shfl_xor(s2, m, 32);
    __shared__ float sh[8];
    if ((threadIdx.x & 31) == 0) sh[threadIdx.x >> 5] = s2;
    __syncthreads();
    s2 = 0.f;
    #pragma unroll
    for (int i = 0; i < 8; i++) s2 += sh[i];
    float r = rsqrtf(s2 * (1.f / DIN) + EPS_);
    __bf16* gr = g + (size_t)row * DIN;
    #pragma unroll
    for (int j = 0; j < 8; j++) {
        int i = threadIdx.x + j * 256;
        gr[i] = (__bf16)(vals[j] * r * rms_w[i]);
    }
}

// ------------------------------ launch -------------------------------------

extern "C" void kernel_launch(void* const* d_in, const int* in_sizes, int n_in,
                              void* d_out, int out_size, void* d_ws, size_t ws_size,
                              hipStream_t stream) {
    const float* x        = (const float*)d_in[0];
    const float* ln_w     = (const float*)d_in[1];
    const float* ln_b     = (const float*)d_in[2];
    const float* in_projw = (const float*)d_in[3];
    const float* conv_w   = (const float*)d_in[4];
    const float* conv_b   = (const float*)d_in[5];
    const float* dt_bias  = (const float*)d_in[6];
    const float* A_log    = (const float*)d_in[7];
    const float* Dv       = (const float*)d_in[8];
    const float* rms_w    = (const float*)d_in[9];
    const float* out_projw= (const float*)d_in[10];
    float* out = (float*)d_out;

    char*  ws  = (char*)d_ws;
    size_t off = 0;
    auto take = [&](size_t bytes) -> void* {
        void* p = ws + off;
        off += (bytes + 255) & ~(size_t)255;
        return p;
    };
    __bf16* ubf  = (__bf16*)take((size_t)ROWS * DM * 2);
    __bf16* w1bf = (__bf16*)take((size_t)DPROJ * DM * 2);
    __bf16* w2bf = (__bf16*)take((size_t)DM * DIN * 2);
    float*  zx   = (float*) take((size_t)ROWS * DPROJ * 4);
    float*  dtb  = (float*) take((size_t)ROWS * NH * 4);
    float*  dAb  = (float*) take((size_t)ROWS * NH * 4);
    float*  xbc  = (float*) take((size_t)ROWS * CONVD * 4);
    float*  yb   = (float*) take((size_t)ROWS * DIN * 4);
    __bf16* gbf  = (__bf16*)take((size_t)ROWS * DIN * 2);

    f32_to_bf16_kernel<<<(DPROJ * DM + 255) / 256, 256, 0, stream>>>(in_projw, w1bf, DPROJ * DM);
    f32_to_bf16_kernel<<<(DM * DIN + 255) / 256, 256, 0, stream>>>(out_projw, w2bf, DM * DIN);

    ln_kernel<<<ROWS, 256, 0, stream>>>(x, ln_w, ln_b, ubf);

    // in_proj: [ROWS,DM] x [DPROJ,DM]^T -> zx [ROWS,DPROJ]
    gemm_bf16_kernel<DM><<<dim3(ROWS / 512, DPROJ / 32), 256, 0, stream>>>(
        ubf, w1bf, zx, nullptr, DPROJ);

    dt_kernel<<<(ROWS * NH + 255) / 256, 256, 0, stream>>>(zx, dt_bias, A_log, dtb, dAb);
    conv_kernel<<<(ROWS * CONVD + 255) / 256, 256, 0, stream>>>(zx, conv_w, conv_b, xbc);

    scan_kernel<<<B_ * NH, 256, 0, stream>>>(xbc, dtb, dAb, Dv, yb);

    gate_kernel<<<ROWS, 256, 0, stream>>>(yb, zx, rms_w, gbf);

    // out_proj: [ROWS,DIN] x [DM,DIN]^T -> out [ROWS,DM], fused +x residual
    gemm_bf16_kernel<DIN><<<dim3(ROWS / 512, DM / 32), 256, 0, stream>>>(
        gbf, w2bf, out, x, DM);
}
